// BinarizedLinearBlock_52072183496926
// MI455X (gfx1250) — matrix operand order repaired and checked by reading, not compiled, not run on hardware
//
#include <hip/hip_runtime.h>
#include <hip/hip_bf16.h>

typedef __attribute__((ext_vector_type(16))) __bf16 v16bf;
typedef __attribute__((ext_vector_type(8)))  __bf16 v8bf;
typedef __attribute__((ext_vector_type(4)))  __bf16 v4bf;
typedef __attribute__((ext_vector_type(8)))  float  v8f;
typedef __attribute__((ext_vector_type(4)))  float  f4;
typedef __attribute__((ext_vector_type(2)))  unsigned int u2;

constexpr int Bsz = 8192;
constexpr int IN  = 2048;
constexpr int OUT = 2048;
constexpr int CHUNKS = 64;
constexpr int ROWS_PER_CHUNK = Bsz / CHUNKS;   // 128

// ---------------- Pass 1: per-column partial sums (deterministic, no atomics)
__global__ __launch_bounds__(256)
void bn_stats_partial(const float* __restrict__ x,
                      float* __restrict__ psum, float* __restrict__ psqs) {
  const int col = blockIdx.x * 256 + threadIdx.x;
  const int r0  = blockIdx.y * ROWS_PER_CHUNK;
  const float* p = x + (size_t)r0 * IN + col;
  float s = 0.f, q = 0.f;
#pragma unroll 4
  for (int r = 0; r < ROWS_PER_CHUNK; ++r) {
    float v = p[(size_t)r * IN];
    s += v; q += v * v;
  }
  psum[blockIdx.y * IN + col] = s;
  psqs[blockIdx.y * IN + col] = q;
}

// ---------------- Pass 2: fold stats into per-column affine (scale, shift)
__global__ __launch_bounds__(256)
void bn_finalize(const float* __restrict__ psum, const float* __restrict__ psqs,
                 const float* __restrict__ gamma, const float* __restrict__ beta,
                 float* __restrict__ scale, float* __restrict__ shift) {
  const int col = blockIdx.x * 256 + threadIdx.x;
  float s = 0.f, q = 0.f;
#pragma unroll 8
  for (int c = 0; c < CHUNKS; ++c) { s += psum[c * IN + col]; q += psqs[c * IN + col]; }
  const float mean = s * (1.0f / Bsz);
  const float var  = q * (1.0f / Bsz) - mean * mean;
  const float rstd = rsqrtf(var + 1e-5f);
  const float g    = gamma[col];
  scale[col] = rstd * g;
  shift[col] = beta[col] - mean * rstd * g;
}

// ---------------- Pass 3: fused BN + binarized GEMM + hardtanh via WMMA bf16
// Block tile 256x128, 8 waves each owning a fat 64x64 (4x4 WMMA tiles):
// 16 v_wmma per 16 ds_load_b128 per k-stage.
constexpr int BM = 256, BN = 128, BK = 32;
constexpr int LDK = BK + 8;        // 40 elems = 80B row stride (16B multiple, bank-spread)
constexpr int NK  = IN / BK;       // 64 k-stages

static __device__ inline v16bf cat16(v8bf lo, v8bf hi) {
  return __builtin_shufflevector(lo, hi, 0,1,2,3,4,5,6,7,8,9,10,11,12,13,14,15);
}

__global__ __launch_bounds__(256)
void bgemm_bin(const float* __restrict__ x, const float* __restrict__ w,
               const float* __restrict__ scale, const float* __restrict__ shift,
               float* __restrict__ out) {
  __shared__ __bf16 As[2][BM][LDK];   // 40 KB
  __shared__ __bf16 Bs[2][BN][LDK];   // 20 KB

  const int tid  = threadIdx.x;
  const int lane = tid & 31;
  const int wid  = tid >> 5;             // 8 waves
  const int wm   = (wid & 3) * 64;       // wave M offset within block tile
  const int wn   = (wid >> 2) * 64;      // wave N offset within block tile
  const int half = lane >> 4;            // WMMA lane-half
  const int l16  = lane & 15;
  const int m0   = blockIdx.y * BM;
  const int n0   = blockIdx.x * BN;

  // global staging: thread -> (row, 4-col chunk); A: 256 rows, B: 128 rows
  const int kk    = (tid & 7) * 4;       // col chunk within BK
  const int rbase = tid >> 3;            // 0..31

  f4 aR[8], bR[4];
  auto loadG = [&](int k0) {
#pragma unroll
    for (int s = 0; s < 8; ++s)
      aR[s] = *(const f4*)(x + (size_t)(m0 + rbase + s * 32) * IN + k0 + kk);
#pragma unroll
    for (int s = 0; s < 4; ++s)
      bR[s] = *(const f4*)(w + (size_t)(n0 + rbase + s * 32) * IN + k0 + kk);
  };

  auto storeL = [&](int buf, int k0) {
    const f4 sc = *(const f4*)(scale + k0 + kk);   // 8 distinct addrs/block: L1 hit
    const f4 sh = *(const f4*)(shift + k0 + kk);
#pragma unroll
    for (int s = 0; s < 8; ++s) {
      v4bf pa;
#pragma unroll
      for (int i = 0; i < 4; ++i)
        pa[i] = (__bf16)(aR[s][i] * sc[i] + sh[i]);      // batchnorm -> bf16
      *(v4bf*)&As[buf][rbase + s * 32][kk] = pa;
    }
#pragma unroll
    for (int s = 0; s < 4; ++s) {
      // sign(w) in bf16 via bit ops: +1 = 0x3F80, -1 = 0xBF80
      const unsigned int b0 = __float_as_uint(bR[s][0]);
      const unsigned int b1 = __float_as_uint(bR[s][1]);
      const unsigned int b2 = __float_as_uint(bR[s][2]);
      const unsigned int b3 = __float_as_uint(bR[s][3]);
      u2 pb;
      pb[0] = 0x3F803F80u | (b1 & 0x80000000u) | ((b0 >> 16) & 0x8000u);
      pb[1] = 0x3F803F80u | (b3 & 0x80000000u) | ((b2 >> 16) & 0x8000u);
      *(u2*)&Bs[buf][rbase + s * 32][kk] = pb;
    }
  };

  v8f acc[4][4];
#pragma unroll
  for (int mt = 0; mt < 4; ++mt)
#pragma unroll
    for (int nt = 0; nt < 4; ++nt)
#pragma unroll
      for (int i = 0; i < 8; ++i) acc[mt][nt][i] = 0.f;

  auto compute = [&](int buf) {
    v16bf aF[4], bF[4];
#pragma unroll
    for (int mt = 0; mt < 4; ++mt) {
      // A 16x32 bf16 layout: lane-half0 K{0..7,16..23}, half1 K{8..15,24..31}
      const __bf16* p = &As[buf][wm + mt * 16 + l16][half * 8];
      v8bf lo = *(const v8bf*)p;
      v8bf hi = *(const v8bf*)(p + 16);
      aF[mt] = cat16(lo, hi);
    }
#pragma unroll
    for (int nt = 0; nt < 4; ++nt) {
      // B 32x16 bf16 layout: lane-half0 K{0..7,16..23}, half1 K{8..15,24..31}
      const __bf16* p = &Bs[buf][wn + nt * 16 + l16][half * 8];
      v8bf lo = *(const v8bf*)p;
      v8bf hi = *(const v8bf*)(p + 16);
      bF[nt] = cat16(lo, hi);
    }
#pragma unroll
    for (int mt = 0; mt < 4; ++mt)
#pragma unroll
      for (int nt = 0; nt < 4; ++nt)
        acc[mt][nt] = __builtin_amdgcn_wmma_f32_16x16x32_bf16(
            false, aF[mt], false, bF[nt], (short)0, acc[mt][nt], false, false);
  };

  // software pipeline: double-buffered LDS, one barrier per stage
  loadG(0);
  storeL(0, 0);
  __syncthreads();
  for (int kt = 0; kt < NK; ++kt) {
    if (kt + 1 < NK) loadG((kt + 1) * BK);
    compute(kt & 1);
    if (kt + 1 < NK) {
      storeL((kt + 1) & 1, (kt + 1) * BK);
      __syncthreads();
    }
  }

  // epilogue: hardtanh + store (lanes 0..15 cover consecutive N -> coalesced)
#pragma unroll
  for (int mt = 0; mt < 4; ++mt) {
#pragma unroll
    for (int nt = 0; nt < 4; ++nt) {
      const int gn = n0 + wn + nt * 16 + l16;
#pragma unroll
      for (int r = 0; r < 8; ++r) {
        const int gm = m0 + wm + mt * 16 + half * 8 + r;
        float v = acc[mt][nt][r];
        v = fminf(fmaxf(v, -1.0f), 1.0f);
        out[(size_t)gm * OUT + gn] = v;
      }
    }
  }
}

extern "C" void kernel_launch(void* const* d_in, const int* in_sizes, int n_in,
                              void* d_out, int out_size, void* d_ws, size_t ws_size,
                              hipStream_t stream) {
  (void)in_sizes; (void)n_in; (void)out_size; (void)ws_size;
  const float* x     = (const float*)d_in[0];
  const float* w     = (const float*)d_in[1];
  const float* gamma = (const float*)d_in[2];
  const float* beta  = (const float*)d_in[3];
  float* out = (float*)d_out;

  float* psum  = (float*)d_ws;              // [64][2048]
  float* psqs  = psum + CHUNKS * IN;        // [64][2048]
  float* scale = psqs + CHUNKS * IN;        // [2048]
  float* shift = scale + IN;                // [2048]

  bn_stats_partial<<<dim3(IN / 256, CHUNKS), 256, 0, stream>>>(x, psum, psqs);
  bn_finalize<<<dim3(IN / 256), 256, 0, stream>>>(psum, psqs, gamma, beta, scale, shift);
  bgemm_bin<<<dim3(OUT / BN, Bsz / BM), 256, 0, stream>>>(x, w, scale, shift, out);
}
